// SingleWindowZigZagAttention_1752346657548
// MI455X (gfx1250) — compile-verified, matrix operand order checked by reading, not run on hardware
//
#include <hip/hip_runtime.h>
#include <hip/hip_bf16.h>

typedef __attribute__((ext_vector_type(16))) _Float16 v16h;
typedef __attribute__((ext_vector_type(8)))  float    v8f;

#define WS_   16
#define NTOK  256       // tokens per window
#define NWIN  256       // windows
#define DIMC  512
#define NHEAD 16
#define HDIM  32
#define NOUT3 1536
#define SCALE 0.17677669529663687f   // 32^-0.5

// ---------------------------------------------------------------------------
// WMMA fragment helpers (CDNA5 16x16x32 f16).
// A-matrix fragment: lane L in [0,16) holds row M=L, K = {0..7}∪{16..23};
// lane L+16 holds row M=L, K = {8..15}∪{24..31}.  B-matrix mirrors with
// lane = column N.  Source layout must be "major index = row (A) / col (B)",
// two 8-half runs (K, K+16) per lane, 16B-aligned (ld multiple of 8 halves).
// ---------------------------------------------------------------------------
__device__ __forceinline__ v16h load_frag(const _Float16* tile, int ldHalves) {
    int lane = threadIdx.x & 31;
    int r    = lane & 15;
    int koff = (lane >> 4) << 3;           // 0 or 8
    const _Float16* p = tile + r * ldHalves + koff;
    union { uint4 u[2]; v16h v; } u;
    u.u[0] = *reinterpret_cast<const uint4*>(p);        // K  = koff .. koff+7
    u.u[1] = *reinterpret_cast<const uint4*>(p + 16);   // K += 16
    return u.v;
}

__device__ __forceinline__ v8f zero8() {
    v8f z;
#pragma unroll
    for (int i = 0; i < 8; ++i) z[i] = 0.0f;
    return z;
}

__device__ __forceinline__ v8f wmma_f16(v16h a, v16h b, v8f c) {
    return __builtin_amdgcn_wmma_f32_16x16x32_f16(
        /*neg_a=*/false, a, /*neg_b=*/false, b,
        /*c_mod=*/(short)0, c, /*reuse_a=*/false, /*reuse_b=*/false);
}

// Async global->LDS copy (gfx1250): 16B per lane, tracked by ASYNCcnt.
// ldsAddr = LDS byte offset (low 32 bits of the flat shared address,
// per the CDNA5 LDS-aperture rule).
__device__ __forceinline__ void async_copy_b128(void* ldsPtr, const void* gPtr) {
    unsigned int  ldsAddr = (unsigned int)(unsigned long long)ldsPtr;
    unsigned long long ga = (unsigned long long)gPtr;
    asm volatile("global_load_async_to_lds_b128 %0, %1, off"
                 :: "v"(ldsAddr), "v"(ga) : "memory");
}
__device__ __forceinline__ void async_wait0() {
    asm volatile("s_wait_asynccnt 0" ::: "memory");
}

// Convert 4 floats -> 4 halves, one 8B LDS store.
__device__ __forceinline__ void cvt4_store(_Float16* dst, float4 f) {
    union { _Float16 h[4]; uint2 u; } p;
    p.h[0] = (_Float16)f.x; p.h[1] = (_Float16)f.y;
    p.h[2] = (_Float16)f.z; p.h[3] = (_Float16)f.w;
    *reinterpret_cast<uint2*>(dst) = p.u;
}

// ---------------------------------------------------------------------------
// Kernel 1: precompute zigzag relative-position bias plane biasM[h][n][m].
// 4 MB, reused by all 256 windows -> stays in the 192 MB L2.
// ---------------------------------------------------------------------------
__global__ __launch_bounds__(256) void bias_init_kernel(
    const float* __restrict__ bias_table, float* __restrict__ biasM) {
    __shared__ int rr[NTOK], cc[NTOK];
    if (threadIdx.x == 0) {
        int r = 0, c = 0;
        for (int i = 0; i < NTOK; ++i) {
            rr[i] = r; cc[i] = c;
            if (((r + c) & 1) == 0) {
                if (c == WS_ - 1) r++;
                else if (r == 0) c++;
                else { r--; c++; }
            } else if (r == WS_ - 1) c++;
            else if (c == 0) r++;
            else { r++; c--; }
        }
    }
    __syncthreads();
    int n = blockIdx.x, m = threadIdx.x;
    int dr = rr[n] - rr[m] + (WS_ - 1);
    int dc = cc[n] - cc[m] + (WS_ - 1);
    int idx = dr * (2 * WS_ - 1) + dc;
#pragma unroll
    for (int h = 0; h < NHEAD; ++h)
        biasM[((size_t)h * NTOK + n) * NTOK + m] = bias_table[idx * NHEAD + h];
}

// ---------------------------------------------------------------------------
// Kernel 2: QKV GEMM  (65536 x 1536 x 512), fp32 in -> fp16 WMMA -> fp16 out.
// Block tile 128(M) x 128(N); 8 waves as 4(M) x 2(N); wave tile 32x64 =
// 2x4 WMMA tiles.  Staging: b128 global loads, packed b64 LDS stores.
// Epilogue scatters q(prescaled)/k row-major and V TRANSPOSED ([bh][hd][n]).
// ---------------------------------------------------------------------------
__global__ __launch_bounds__(256) void qkv_gemm_kernel(
    const float* __restrict__ X, const float* __restrict__ W,
    const float* __restrict__ bias,
    _Float16* __restrict__ qws, _Float16* __restrict__ kws,
    _Float16* __restrict__ vtws) {
    constexpr int LDA = 48, LDB = 48;
    __shared__ _Float16 As[128 * LDA];
    __shared__ _Float16 Bs[128 * LDB];   // Bs[n][k] (column-major of W)

    int tid = threadIdx.x;
    int rowBase = blockIdx.y * 128;
    int colBase = blockIdx.x * 128;
    int wid = tid >> 5;
    int wm = wid & 3, wn = wid >> 2;

    v8f acc[2][4];
#pragma unroll
    for (int i = 0; i < 2; ++i)
#pragma unroll
        for (int j = 0; j < 4; ++j) acc[i][j] = zero8();

    for (int kt = 0; kt < DIMC / 32; ++kt) {
        // A: 128x32 fp32 -> fp16; 1024 float4 chunks, 4 per thread
#pragma unroll
        for (int t = 0; t < 4; ++t) {
            int j = tid + t * 256;
            int r = j >> 3, c4 = (j & 7) * 4;
            const float* src = &X[(size_t)(rowBase + r) * DIMC + kt * 32 + c4];
            cvt4_store(&As[r * LDA + c4], *reinterpret_cast<const float4*>(src));
        }
        // branch-free prefetch of next A tile (wraps at the end; harmless)
        {
            int j = tid;
            int r = j >> 3, c4 = (j & 7) * 4;
            int ktn = (kt + 1) & (DIMC / 32 - 1);
            __builtin_prefetch(&X[(size_t)(rowBase + r) * DIMC + ktn * 32 + c4], 0, 1);
        }
        // B: 32x128 fp32 -> Bs[n][k]; 4 coalesced b32 loads + 1 b64 store
#pragma unroll
        for (int t = 0; t < 4; ++t) {
            int j = tid + t * 256;
            int n = j & 127, c4 = (j >> 7) * 4;
            float4 f;
            f.x = W[(size_t)(kt * 32 + c4 + 0) * NOUT3 + colBase + n];
            f.y = W[(size_t)(kt * 32 + c4 + 1) * NOUT3 + colBase + n];
            f.z = W[(size_t)(kt * 32 + c4 + 2) * NOUT3 + colBase + n];
            f.w = W[(size_t)(kt * 32 + c4 + 3) * NOUT3 + colBase + n];
            cvt4_store(&Bs[n * LDB + c4], f);
        }
        __syncthreads();
        v16h bf[4];
#pragma unroll
        for (int ni = 0; ni < 4; ++ni)
            bf[ni] = load_frag(&Bs[(wn * 64 + ni * 16) * LDB], LDB);
#pragma unroll
        for (int mi = 0; mi < 2; ++mi) {
            v16h a = load_frag(&As[(wm * 32 + mi * 16) * LDA], LDA);
#pragma unroll
            for (int ni = 0; ni < 4; ++ni)
                acc[mi][ni] = wmma_f16(a, bf[ni], acc[mi][ni]);
        }
        __syncthreads();
    }

    int lane = tid & 31;
    int rOff = (lane >> 4) << 3;
    int cOff = lane & 15;
#pragma unroll
    for (int mi = 0; mi < 2; ++mi)
#pragma unroll
        for (int ni = 0; ni < 4; ++ni)
#pragma unroll
            for (int v = 0; v < 8; ++v) {
                int row = rowBase + wm * 32 + mi * 16 + v + rOff;
                int col = colBase + wn * 64 + ni * 16 + cOff;
                float val = acc[mi][ni][v] + bias[col];
                int part = col >> 9;           // 0=q 1=k 2=v
                int rem  = col & 511;
                int h = rem >> 5, hd = rem & 31;
                int b = row >> 8, n = row & 255;
                if (part == 0) {
                    size_t dst = (((size_t)(b * NHEAD + h)) * NTOK + n) * HDIM + hd;
                    qws[dst] = (_Float16)(val * SCALE);
                } else if (part == 1) {
                    size_t dst = (((size_t)(b * NHEAD + h)) * NTOK + n) * HDIM + hd;
                    kws[dst] = (_Float16)val;
                } else {
                    // V stored transposed: [bh][hd][n]
                    size_t dst = (((size_t)(b * NHEAD + h)) * HDIM + hd) * NTOK + n;
                    vtws[dst] = (_Float16)val;
                }
            }
}

// ---------------------------------------------------------------------------
// Kernel 3: attention for one (window, head) per workgroup; scores never
// touch HBM.  LDS: Q,K row-major (ld 48), V transposed (ld 272),
// S/P fp16 (ld 272)  -> ~201 KB of the 320 KB WGP LDS.
// ALL operands staged with global_load_async_to_lds_b128 (ASYNCcnt path).
// ---------------------------------------------------------------------------
#define LDQ  48
#define LDSS 272
#define LDV  272
#define ATTN_LDS_BYTES ((256 * LDQ * 2 + 32 * LDV + 256 * LDSS) * 2)

__global__ __launch_bounds__(256) void attn_kernel(
    const _Float16* __restrict__ qws, const _Float16* __restrict__ kws,
    const _Float16* __restrict__ vtws, const float* __restrict__ biasM,
    _Float16* __restrict__ aout) {
    extern __shared__ _Float16 smem[];
    _Float16* Qs = smem;                       // 256 x 48
    _Float16* Ks = Qs + 256 * LDQ;             // 256 x 48
    _Float16* Vt = Ks + 256 * LDQ;             // 32  x 272 (transposed)
    _Float16* Ss = Vt + 32 * LDV;              // 256 x 272

    int tid = threadIdx.x;
    int bh  = blockIdx.x;
    int b   = bh >> 4, h = bh & 15;
    const _Float16* qg  = qws  + (size_t)bh * NTOK * HDIM;
    const _Float16* kg  = kws  + (size_t)bh * NTOK * HDIM;
    const _Float16* vtg = vtws + (size_t)bh * NTOK * HDIM;  // [hd][n]

    // Q/K: 1024 16B chunks each (row n = j>>2, halves d = (j&3)*8)
#pragma unroll
    for (int t = 0; t < 4; ++t) {
        int j = tid + t * 256;
        int n = j >> 2, d = (j & 3) * 8;
        async_copy_b128(&Qs[n * LDQ + d], qg + j * 8);
        async_copy_b128(&Ks[n * LDQ + d], kg + j * 8);
    }
    // Vt: already transposed in global; 1024 chunks (row hd = j>>5)
#pragma unroll
    for (int t = 0; t < 4; ++t) {
        int j = tid + t * 256;
        int hd = j >> 5, n0 = (j & 31) * 8;
        async_copy_b128(&Vt[hd * LDV + n0], vtg + j * 8);
    }
    async_wait0();
    __syncthreads();

    int wid = tid >> 5, lane = tid & 31;
    int rOff = (lane >> 4) << 3;
    int cOff = lane & 15;
    const float* bp = biasM + (size_t)h * NTOK * NTOK;

    // S = Q K^T + bias  (one 16x16x32 WMMA per tile, HD==32==K)
#pragma unroll
    for (int mi = 0; mi < 2; ++mi) {
        int mt = wid * 2 + mi;
        v16h a = load_frag(&Qs[(mt * 16) * LDQ], LDQ);
        for (int nt = 0; nt < 16; ++nt) {
            v16h bf = load_frag(&Ks[(nt * 16) * LDQ], LDQ);
            v8f c = zero8();
            c = wmma_f16(a, bf, c);
#pragma unroll
            for (int v = 0; v < 8; ++v) {
                int rr = mt * 16 + v + rOff;
                int cc = nt * 16 + cOff;
                Ss[rr * LDSS + cc] = (_Float16)(c[v] + bp[(size_t)rr * NTOK + cc]);
            }
        }
    }
    __syncthreads();

    // row softmax (thread == row), vectorized 8 halves per LDS op
    {
        _Float16* srow = Ss + tid * LDSS;
        union V8 { uint4 u; _Float16 h[8]; };
        float mx = -1e30f;
        for (int j = 0; j < NTOK; j += 8) {
            V8 w; w.u = *(const uint4*)(srow + j);
#pragma unroll
            for (int k = 0; k < 8; ++k) mx = fmaxf(mx, (float)w.h[k]);
        }
        float sum = 0.0f;
        for (int j = 0; j < NTOK; j += 8) {
            V8 w; w.u = *(const uint4*)(srow + j);
#pragma unroll
            for (int k = 0; k < 8; ++k) {
                float e = __expf((float)w.h[k] - mx);
                sum += e;
                w.h[k] = (_Float16)e;
            }
            *(uint4*)(srow + j) = w.u;
        }
        float inv = 1.0f / sum;
        for (int j = 0; j < NTOK; j += 8) {
            V8 w; w.u = *(const uint4*)(srow + j);
#pragma unroll
            for (int k = 0; k < 8; ++k) w.h[k] = (_Float16)((float)w.h[k] * inv);
            *(uint4*)(srow + j) = w.u;
        }
    }
    __syncthreads();

    // O = P V : 16 x 2 output tiles, 4 per wave, 8 K-steps of 32
#pragma unroll
    for (int q = 0; q < 4; ++q) {
        int tt = wid * 4 + q;
        int mt = tt >> 1, ht = tt & 1;
        v8f c = zero8();
#pragma unroll
        for (int kb = 0; kb < 8; ++kb) {
            v16h a  = load_frag(&Ss[(mt * 16) * LDSS + kb * 32], LDSS);
            v16h bf = load_frag(&Vt[(ht * 16) * LDV + kb * 32], LDV);
            c = wmma_f16(a, bf, c);
        }
#pragma unroll
        for (int v = 0; v < 8; ++v) {
            int n  = mt * 16 + v + rOff;
            int hd = ht * 16 + cOff;
            aout[((size_t)(b * NTOK + n)) * DIMC + h * HDIM + hd] = (_Float16)c[v];
        }
    }
}

// ---------------------------------------------------------------------------
// Kernel 4: output projection GEMM (65536 x 512 x 512), fp16 A, fp32 out.
// A tile is a pure fp16 copy -> staged with async global->LDS b128.
// ---------------------------------------------------------------------------
__global__ __launch_bounds__(256) void proj_gemm_kernel(
    const _Float16* __restrict__ X, const float* __restrict__ W,
    const float* __restrict__ bias, float* __restrict__ out) {
    constexpr int LDA = 48, LDB = 48;
    __shared__ _Float16 As[128 * LDA];
    __shared__ _Float16 Bs[128 * LDB];

    int tid = threadIdx.x;
    int rowBase = blockIdx.y * 128;
    int colBase = blockIdx.x * 128;
    int wid = tid >> 5;
    int wm = wid & 3, wn = wid >> 2;

    v8f acc[2][4];
#pragma unroll
    for (int i = 0; i < 2; ++i)
#pragma unroll
        for (int j = 0; j < 4; ++j) acc[i][j] = zero8();

    for (int kt = 0; kt < DIMC / 32; ++kt) {
        // A: 128x32 fp16 tile, 512 16B chunks, async copy (2 per thread)
#pragma unroll
        for (int t = 0; t < 2; ++t) {
            int j = tid + t * 256;
            int r = j >> 2, c = (j & 3) * 8;
            async_copy_b128(&As[r * LDA + c],
                            &X[(size_t)(rowBase + r) * DIMC + kt * 32 + c]);
        }
        // B: 32x128 fp32 -> Bs[n][k]; 4 coalesced b32 loads + 1 b64 store
#pragma unroll
        for (int t = 0; t < 4; ++t) {
            int j = tid + t * 256;
            int n = j & 127, c4 = (j >> 7) * 4;
            float4 f;
            f.x = W[(size_t)(kt * 32 + c4 + 0) * DIMC + colBase + n];
            f.y = W[(size_t)(kt * 32 + c4 + 1) * DIMC + colBase + n];
            f.z = W[(size_t)(kt * 32 + c4 + 2) * DIMC + colBase + n];
            f.w = W[(size_t)(kt * 32 + c4 + 3) * DIMC + colBase + n];
            cvt4_store(&Bs[n * LDB + c4], f);
        }
        async_wait0();
        __syncthreads();
        v16h bf[4];
#pragma unroll
        for (int ni = 0; ni < 4; ++ni)
            bf[ni] = load_frag(&Bs[(wn * 64 + ni * 16) * LDB], LDB);
#pragma unroll
        for (int mi = 0; mi < 2; ++mi) {
            v16h a = load_frag(&As[(wm * 32 + mi * 16) * LDA], LDA);
#pragma unroll
            for (int ni = 0; ni < 4; ++ni)
                acc[mi][ni] = wmma_f16(a, bf[ni], acc[mi][ni]);
        }
        __syncthreads();
    }

    int lane = tid & 31;
    int rOff = (lane >> 4) << 3;
    int cOff = lane & 15;
#pragma unroll
    for (int mi = 0; mi < 2; ++mi)
#pragma unroll
        for (int ni = 0; ni < 4; ++ni)
#pragma unroll
            for (int v = 0; v < 8; ++v) {
                int row = rowBase + wm * 32 + mi * 16 + v + rOff;
                int col = colBase + wn * 64 + ni * 16 + cOff;
                out[(size_t)row * DIMC + col] = acc[mi][ni][v] + bias[col];
            }
}

// ---------------------------------------------------------------------------
// Launch: bias plane -> QKV GEMM -> per-(b,h) attention -> projection.
// Workspace layout (needs 260 MiB): q 64MiB | k 64MiB | vT 64MiB |
//                                   attn_out 64MiB | biasM 4MiB
// ---------------------------------------------------------------------------
extern "C" void kernel_launch(void* const* d_in, const int* in_sizes, int n_in,
                              void* d_out, int out_size, void* d_ws, size_t ws_size,
                              hipStream_t stream) {
    (void)in_sizes; (void)n_in; (void)out_size; (void)ws_size;
    const float* inputs     = (const float*)d_in[0];
    const float* qkv_w      = (const float*)d_in[1];
    const float* qkv_b      = (const float*)d_in[2];
    const float* proj_w     = (const float*)d_in[3];
    const float* proj_b     = (const float*)d_in[4];
    const float* bias_table = (const float*)d_in[5];
    float* out = (float*)d_out;

    char* ws = (char*)d_ws;
    const size_t MB64 = (size_t)64 << 20;
    _Float16* qws   = (_Float16*)(ws);
    _Float16* kws   = (_Float16*)(ws + 1 * MB64);
    _Float16* vtws  = (_Float16*)(ws + 2 * MB64);
    _Float16* aout  = (_Float16*)(ws + 3 * MB64);
    float*    biasM = (float*)   (ws + 4 * MB64);

    bias_init_kernel<<<NTOK, 256, 0, stream>>>(bias_table, biasM);

    qkv_gemm_kernel<<<dim3(NOUT3 / 128, (NWIN * NTOK) / 128), 256, 0, stream>>>(
        inputs, qkv_w, qkv_b, qws, kws, vtws);

    attn_kernel<<<NWIN * NHEAD, 256, ATTN_LDS_BYTES, stream>>>(
        qws, kws, vtws, biasM, aout);

    proj_gemm_kernel<<<dim3(DIMC / 128, (NWIN * NTOK) / 128), 256, 0, stream>>>(
        aout, proj_w, proj_b, out);
}
